// _GeoBase_40699110097485
// MI455X (gfx1250) — matrix operand order repaired and checked by reading, not compiled, not run on hardware
//
#include <hip/hip_runtime.h>
#include <hip/hip_bf16.h>
#include <math.h>

// ---------------- problem constants (match reference) ----------------
#define GN    50000   // nodes
#define GE    800000  // edges
#define GIN   32      // in_channels
#define GHID  64      // hidden
#define GHEADS 4
#define GC    16      // per-head channels
#define GEDIM 16      // edge_dim
#define GL    3       // layers
#define GOUT  3

typedef __attribute__((ext_vector_type(16))) _Float16 v16h;
typedef __attribute__((ext_vector_type(8)))  _Float16 v8h;
typedef __attribute__((ext_vector_type(8)))  float    v8f;

// ---------------- helpers ----------------
__device__ __forceinline__ float atomicMaxFloat(float* addr, float value) {
    // sign-aware bit trick: int max for >=0, uint min for <0
    if (value >= 0.0f)
        return __int_as_float(atomicMax((int*)addr, __float_as_int(value)));
    else
        return __uint_as_float(atomicMin((unsigned int*)addr, __float_as_uint(value)));
}

__global__ void fill_kernel(float* __restrict__ p, float val, int cnt) {
    int i = blockIdx.x * 256 + threadIdx.x;
    if (i < cnt) p[i] = val;
}

// ---------------- WMMA GEMM: Y[mat] = act(A[MxKK] @ W[mat][KKx64] + b[mat]) ----------------
// block = 256 threads = 8 wave32s; block covers 32 rows x 64 cols (2 row tiles x 4 col tiles)
// Weight panel staged in LDS in B-fragment-packed layout:
//   sWp[col*72 + seg*16 + j]  (halfs), seg = K-segment of 16, col stride 72 (144B, 16B-aligned,
//   bank-spread) so each lane's 16-half B fragment is two contiguous ds_load_b128's.
struct Gemm4 {
    const float* W[4];
    const float* Bv[4];
    float*       Y[4];
};

#define WCOLSTRIDE 72

template <int KK>
__global__ void __launch_bounds__(256)
gemm_wmma_kernel(const float* __restrict__ A, int M, Gemm4 g, int relu_act) {
    const int mat = blockIdx.y;
    const float* __restrict__ W    = g.W[mat];
    const float* __restrict__ bias = g.Bv[mat];
    float* __restrict__ Y          = g.Y[mat];

    __shared__ _Float16 sWp[64 * WCOLSTRIDE];   // 9216 B

    const int tid  = threadIdx.x;
    const int wave = tid >> 5;
    const int lane = tid & 31;
    const int rowTile = wave >> 2;        // 0..1
    const int colTile = wave & 3;         // 0..3
    const int row0    = blockIdx.x * 32 + rowTile * 16;
    const int n0      = colTile * 16;

    const int arow  = row0 + (lane & 15); // A row this lane feeds
    const int khalf = lane >> 4;          // K half this lane holds (16 contiguous k per half)
    const int col   = n0 + (lane & 15);   // C/D column this lane holds
    const bool fullTile = (row0 + 16 <= M);   // only the last row-block is partial

    // pull this lane's A row toward the WGP while weights stage through LDS
    if (arow < M) __builtin_prefetch(A + (size_t)arow * KK, 0, 0);

    // stage weights: coalesced global read, fragment-packed LDS write
    for (int idx = tid; idx < KK * 64; idx += 256) {
        const int wc = idx & 63;          // column
        const int wk = idx >> 6;          // k index
        sWp[wc * WCOLSTRIDE + (wk >> 4) * 16 + (wk & 15)] = (_Float16)W[idx];
    }
    __syncthreads();

    v8f c = {};
    #pragma unroll
    for (int kc = 0; kc < KK; kc += 32) {
        // A tile 16x32 f16: lane holds K = kc + khalf*16 .. +15
        v16h a;
        if (fullTile || arow < M) {
            const float* ap = A + (size_t)arow * KK + kc + khalf * 16;
            #pragma unroll
            for (int j = 0; j < 8; ++j) a[j] = (_Float16)ap[j];
            #pragma unroll
            for (int j = 0; j < 8; ++j) a[8 + j] = (_Float16)ap[8 + j];
        } else {
            #pragma unroll
            for (int j = 0; j < 16; ++j) a[j] = (_Float16)0.0f;
        }
        // B tile 32x16 f16: lane = column, elements = K kc + khalf*16 .. +15
        // packed layout -> two contiguous 16B LDS loads
        const v8h* bp = (const v8h*)(sWp + col * WCOLSTRIDE + (kc >> 4) * 16 + khalf * 16);
        const v8h blo = bp[0];
        const v8h bhi = bp[1];
        v16h b;
        #pragma unroll
        for (int j = 0; j < 8; ++j) { b[j] = blo[j]; b[8 + j] = bhi[j]; }

        c = __builtin_amdgcn_wmma_f32_16x16x32_f16(
                /*neg_a=*/false, a, /*neg_b=*/false, b,
                /*c_mod=*/(short)0, c, /*reuse_a=*/false, /*reuse_b=*/false);
    }

    const float bb = bias[col];
    if (fullTile) {
        // straight-line epilogue: each store is 64B contiguous per half-wave
        #pragma unroll
        for (int r = 0; r < 8; ++r) {
            const int m = row0 + khalf * 8 + r;
            float v = c[r] + bb;
            if (relu_act) v = fmaxf(v, 0.0f);
            Y[(size_t)m * 64 + col] = v;
        }
    } else {
        #pragma unroll
        for (int r = 0; r < 8; ++r) {
            const int m = row0 + khalf * 8 + r;
            if (m < M) {
                float v = c[r] + bb;
                if (relu_act) v = fmaxf(v, 0.0f);
                Y[(size_t)m * 64 + col] = v;
            }
        }
    }
}

// ---------------- edge pass 1: alpha = scale * dot(q[dst], k[src]+e), seg-max ----------------
// 64 threads per edge (one per channel), 4 edges per 256-thread block.
// e_w / e_b staged in LDS (conflict-free: ch stride-1 across lanes).
__global__ void __launch_bounds__(256)
edge_alpha_kernel(const long long* __restrict__ ei, const float* __restrict__ edge_attr,
                  const float* __restrict__ ew, const float* __restrict__ ebias,
                  const float* __restrict__ q, const float* __restrict__ k,
                  float* __restrict__ alpha, float* __restrict__ mbuf, int Ecnt) {
    __shared__ float sEW[GEDIM * GHID];   // 4 KB
    __shared__ float sEB[GHID];
    const int tid = threadIdx.x;
    for (int idx = tid; idx < GEDIM * GHID; idx += 256) sEW[idx] = ew[idx];
    if (tid < GHID) sEB[tid] = ebias[tid];
    __syncthreads();

    const int e = blockIdx.x * 4 + (tid >> 6);
    if (e >= Ecnt) return;
    const int ch   = tid & 63;
    const int head = ch >> 4;
    const int src  = (int)ei[e];
    const int dst  = (int)ei[Ecnt + e];

    // e_ch = edge_attr[e,:] . ew[:,ch] + eb[ch]   (edge_attr row via 4x b128)
    const float4* ea4 = (const float4*)(edge_attr + (size_t)e * GEDIM);
    float ec = sEB[ch];
    #pragma unroll
    for (int q4 = 0; q4 < 4; ++q4) {
        const float4 v4 = ea4[q4];
        ec = fmaf(v4.x, sEW[(q4 * 4 + 0) * GHID + ch], ec);
        ec = fmaf(v4.y, sEW[(q4 * 4 + 1) * GHID + ch], ec);
        ec = fmaf(v4.z, sEW[(q4 * 4 + 2) * GHID + ch], ec);
        ec = fmaf(v4.w, sEW[(q4 * 4 + 3) * GHID + ch], ec);
    }

    float p = q[(size_t)dst * GHID + ch] * (k[(size_t)src * GHID + ch] + ec);
    #pragma unroll
    for (int m = 1; m < 16; m <<= 1) p += __shfl_xor(p, m, 32);

    if ((ch & 15) == 0) {
        const float a = p * 0.25f;  // 1/sqrt(C=16)
        alpha[(size_t)e * GHEADS + head] = a;
        atomicMaxFloat(&mbuf[(size_t)dst * GHEADS + head], a);
    }
}

// ---------------- edge pass 2: ex = exp(alpha - m[dst]); seg-sum den ----------------
__global__ void __launch_bounds__(256)
edge_exp_kernel(const long long* __restrict__ ei, float* __restrict__ alpha,
                const float* __restrict__ mbuf, float* __restrict__ den, int Ecnt) {
    const int idx = blockIdx.x * 256 + threadIdx.x;
    if (idx >= Ecnt * GHEADS) return;
    const int e = idx >> 2, head = idx & 3;
    const int dst = (int)ei[Ecnt + e];
    const float ex = __expf(alpha[idx] - mbuf[(size_t)dst * GHEADS + head]);
    alpha[idx] = ex;
    atomicAdd(&den[(size_t)dst * GHEADS + head], ex);
}

// ---------------- edge pass 3: agg[dst] += a * (v[src] + e) ----------------
__global__ void __launch_bounds__(256)
edge_msg_kernel(const long long* __restrict__ ei, const float* __restrict__ edge_attr,
                const float* __restrict__ ew, const float* __restrict__ ebias,
                const float* __restrict__ v, const float* __restrict__ alpha,
                const float* __restrict__ den, float* __restrict__ agg, int Ecnt) {
    __shared__ float sEW[GEDIM * GHID];
    __shared__ float sEB[GHID];
    const int tid = threadIdx.x;
    for (int idx = tid; idx < GEDIM * GHID; idx += 256) sEW[idx] = ew[idx];
    if (tid < GHID) sEB[tid] = ebias[tid];
    __syncthreads();

    const int e = blockIdx.x * 4 + (tid >> 6);
    if (e >= Ecnt) return;
    const int ch   = tid & 63;
    const int head = ch >> 4;
    const int src  = (int)ei[e];
    const int dst  = (int)ei[Ecnt + e];

    const float4* ea4 = (const float4*)(edge_attr + (size_t)e * GEDIM);
    float ec = sEB[ch];
    #pragma unroll
    for (int q4 = 0; q4 < 4; ++q4) {
        const float4 v4 = ea4[q4];
        ec = fmaf(v4.x, sEW[(q4 * 4 + 0) * GHID + ch], ec);
        ec = fmaf(v4.y, sEW[(q4 * 4 + 1) * GHID + ch], ec);
        ec = fmaf(v4.z, sEW[(q4 * 4 + 2) * GHID + ch], ec);
        ec = fmaf(v4.w, sEW[(q4 * 4 + 3) * GHID + ch], ec);
    }

    const float a = alpha[(size_t)e * GHEADS + head] /
                    (den[(size_t)dst * GHEADS + head] + 1e-16f);
    atomicAdd(&agg[(size_t)dst * GHID + ch],
              (v[(size_t)src * GHID + ch] + ec) * a);
}

// ---------------- node epilogue: beta gate + residual + LayerNorm + ReLU ----------------
// one wave32 per node, 2 channels per lane; 8 nodes per 256-thread block
__global__ void __launch_bounds__(256)
node_update_kernel(float* __restrict__ h, const float* __restrict__ agg,
                   const float* __restrict__ xr, const float* __restrict__ betaw,
                   const float* __restrict__ lng, const float* __restrict__ lnb, int Nn) {
    const int n = blockIdx.x * 8 + (threadIdx.x >> 5);
    if (n >= Nn) return;
    const int lane = threadIdx.x & 31;
    const int c0 = lane, c1 = lane + 32;
    const size_t base = (size_t)n * GHID;

    const float o0 = agg[base + c0], o1 = agg[base + c1];
    const float x0 = xr[base + c0],  x1 = xr[base + c1];

    // g = sigmoid([out, xr, out-xr] . beta_w)
    float t = o0 * betaw[c0]       + o1 * betaw[c1]
            + x0 * betaw[64 + c0]  + x1 * betaw[64 + c1]
            + (o0 - x0) * betaw[128 + c0] + (o1 - x1) * betaw[128 + c1];
    #pragma unroll
    for (int m = 1; m < 32; m <<= 1) t += __shfl_xor(t, m, 32);
    const float g = 1.0f / (1.0f + __expf(-t));

    float z0 = h[base + c0] + g * x0 + (1.0f - g) * o0;
    float z1 = h[base + c1] + g * x1 + (1.0f - g) * o1;

    float s = z0 + z1;
    #pragma unroll
    for (int m = 1; m < 32; m <<= 1) s += __shfl_xor(s, m, 32);
    const float mu = s * (1.0f / 64.0f);
    const float d0 = z0 - mu, d1 = z1 - mu;
    float vs = d0 * d0 + d1 * d1;
    #pragma unroll
    for (int m = 1; m < 32; m <<= 1) vs += __shfl_xor(vs, m, 32);
    const float rs = rsqrtf(vs * (1.0f / 64.0f) + 1e-5f);

    h[base + c0] = fmaxf(d0 * rs * lng[c0] + lnb[c0], 0.0f);
    h[base + c1] = fmaxf(d1 * rs * lng[c1] + lnb[c1], 0.0f);
}

// ---------------- final tiny GEMV: out[N,3] = h1 @ out2_w + b ----------------
__global__ void __launch_bounds__(256)
out2_kernel(const float* __restrict__ h1, const float* __restrict__ w,
            const float* __restrict__ b, float* __restrict__ out, int Nn) {
    const int n = blockIdx.x * 256 + threadIdx.x;
    if (n >= Nn) return;
    float a0 = b[0], a1 = b[1], a2 = b[2];
    const float* hr = h1 + (size_t)n * GHID;
    #pragma unroll
    for (int j = 0; j < GHID; ++j) {
        const float hv = hr[j];
        a0 = fmaf(hv, w[j * GOUT + 0], a0);
        a1 = fmaf(hv, w[j * GOUT + 1], a1);
        a2 = fmaf(hv, w[j * GOUT + 2], a2);
    }
    out[(size_t)n * GOUT + 0] = a0;
    out[(size_t)n * GOUT + 1] = a1;
    out[(size_t)n * GOUT + 2] = a2;
}

// ---------------- launcher ----------------
extern "C" void kernel_launch(void* const* d_in, const int* in_sizes, int n_in,
                              void* d_out, int out_size, void* d_ws, size_t ws_size,
                              hipStream_t stream) {
    (void)in_sizes; (void)n_in; (void)out_size; (void)ws_size;

    const float*      x      = (const float*)d_in[0];
    const long long*  ei     = (const long long*)d_in[1];  // int64 (2,E)
    const float*      eattr  = (const float*)d_in[2];
    const float*      in_w   = (const float*)d_in[3];
    const float*      in_b   = (const float*)d_in[4];
    const float*      q_w    = (const float*)d_in[5];
    const float*      q_b    = (const float*)d_in[6];
    const float*      k_w    = (const float*)d_in[7];
    const float*      k_b    = (const float*)d_in[8];
    const float*      v_w    = (const float*)d_in[9];
    const float*      v_b    = (const float*)d_in[10];
    const float*      e_w    = (const float*)d_in[11];
    const float*      e_b    = (const float*)d_in[12];
    const float*      sk_w   = (const float*)d_in[13];
    const float*      sk_b   = (const float*)d_in[14];
    const float*      beta_w = (const float*)d_in[15];
    const float*      ln_g   = (const float*)d_in[16];
    const float*      ln_b   = (const float*)d_in[17];
    const float*      o1_w   = (const float*)d_in[18];
    const float*      o1_b   = (const float*)d_in[19];
    const float*      o2_w   = (const float*)d_in[20];
    const float*      o2_b   = (const float*)d_in[21];
    float*            out    = (float*)d_out;

    // workspace layout (floats)
    const size_t N64 = (size_t)GN * GHID;     // 3.2M floats
    float* ws    = (float*)d_ws;
    float* h     = ws;
    float* qb    = ws + 1 * N64;
    float* kb    = ws + 2 * N64;
    float* vb    = ws + 3 * N64;
    float* xrb   = ws + 4 * N64;
    float* aggb  = ws + 5 * N64;
    float* alpha = ws + 6 * N64;                       // E*4 floats
    float* mbuf  = alpha + (size_t)GE * GHEADS;        // N*4
    float* denb  = mbuf + (size_t)GN * GHEADS;         // N*4
    float* h1    = qb;  // reuse q buffer for output-MLP hidden

    const dim3 blk(256);
    const dim3 gemmG1((GN + 31) / 32, 1);
    const dim3 gemmG4((GN + 31) / 32, 4);
    const dim3 edgeG((GE + 3) / 4);
    const dim3 expG((GE * GHEADS + 255) / 256);
    const dim3 nodeG((GN + 7) / 8);

    // input projection: h = x @ in_w + in_b   (K = 32)
    {
        Gemm4 g = {};
        g.W[0] = in_w; g.Bv[0] = in_b; g.Y[0] = h;
        gemm_wmma_kernel<GIN><<<gemmG1, blk, 0, stream>>>(x, GN, g, 0);
    }

    for (int l = 0; l < GL; ++l) {
        // reset segment buffers
        fill_kernel<<<(GN * GHEADS + 255) / 256, blk, 0, stream>>>(mbuf, -3.402823466e38f, GN * GHEADS);
        fill_kernel<<<(GN * GHEADS + 255) / 256, blk, 0, stream>>>(denb, 0.0f, GN * GHEADS);
        fill_kernel<<<((int)N64 + 255) / 256, blk, 0, stream>>>(aggb, 0.0f, (int)N64);

        // q,k,v,skip = h @ {W} + b  (fused 4-matrix batch, K = 64)
        {
            Gemm4 g;
            g.W[0] = q_w  + (size_t)l * GHID * GHID; g.Bv[0] = q_b  + (size_t)l * GHID; g.Y[0] = qb;
            g.W[1] = k_w  + (size_t)l * GHID * GHID; g.Bv[1] = k_b  + (size_t)l * GHID; g.Y[1] = kb;
            g.W[2] = v_w  + (size_t)l * GHID * GHID; g.Bv[2] = v_b  + (size_t)l * GHID; g.Y[2] = vb;
            g.W[3] = sk_w + (size_t)l * GHID * GHID; g.Bv[3] = sk_b + (size_t)l * GHID; g.Y[3] = xrb;
            gemm_wmma_kernel<GHID><<<gemmG4, blk, 0, stream>>>(h, GN, g, 0);
        }

        const float* ewl = e_w + (size_t)l * GEDIM * GHID;
        const float* ebl = e_b + (size_t)l * GHID;

        edge_alpha_kernel<<<edgeG, blk, 0, stream>>>(ei, eattr, ewl, ebl, qb, kb, alpha, mbuf, GE);
        edge_exp_kernel<<<expG, blk, 0, stream>>>(ei, alpha, mbuf, denb, GE);
        edge_msg_kernel<<<edgeG, blk, 0, stream>>>(ei, eattr, ewl, ebl, vb, alpha, denb, aggb, GE);

        node_update_kernel<<<nodeG, blk, 0, stream>>>(h, aggb, xrb,
                beta_w + (size_t)l * 3 * GHID,
                ln_g + (size_t)l * GHID, ln_b + (size_t)l * GHID, GN);
    }

    // output MLP
    {
        Gemm4 g = {};
        g.W[0] = o1_w; g.Bv[0] = o1_b; g.Y[0] = h1;
        gemm_wmma_kernel<GHID><<<gemmG1, blk, 0, stream>>>(h, GN, g, 1 /*relu*/);
    }
    out2_kernel<<<(GN + 255) / 256, blk, 0, stream>>>(h1, o2_w, o2_b, out, GN);
}
